// Int8Linear_4733053960568
// MI455X (gfx1250) — matrix operand order repaired and checked by reading, not compiled, not run on hardware
//

#include <hip/hip_runtime.h>
#include <hip/hip_bf16.h>

// Int8-weight linear: out[M,N] = x[M,K] @ (w_int8[N,K] * scale[N])^T + bias[N]
// M = 4*2048 = 8192, N = 11008, K = 4096.
//
// Memory-bound problem (~650MB of mandatory HBM traffic vs 0.74 TFLOP).
// Plan: (1) dequant int32-stored-int8 weights -> f16 once (172MB -> 86MB),
//       (2) convert x fp32 -> f16 once (128MB -> 64MB) so both GEMM operands
//           (150MB) live in the 192MB L2 for all tile re-reads,
//       (3) WMMA f16 GEMM with fp32 accumulate, 128x128x32 tiles,
//           double-buffered LDS fed by gfx1250 async direct-to-LDS loads
//           (ASYNCcnt) when available, register staging otherwise.

typedef _Float16 v16h __attribute__((ext_vector_type(16)));
typedef _Float16 v8h  __attribute__((ext_vector_type(8)));
typedef _Float16 v4h  __attribute__((ext_vector_type(4)));
typedef float    v8f  __attribute__((ext_vector_type(8)));
typedef float    v4f  __attribute__((ext_vector_type(4)));
typedef int      v4i  __attribute__((ext_vector_type(4)));

#define M_TOT 8192
#define N_TOT 11008
#define K_TOT 4096
#define BLK_M 128
#define BLK_N 128
#define BLK_K 32
#define LDSROW 40   // 32 f16 payload + 8 f16 pad -> 80B row stride, conflict-free

#if defined(__has_builtin)
#if __has_builtin(__builtin_amdgcn_global_load_async_to_lds_b128) && \
    __has_builtin(__builtin_amdgcn_s_wait_asynccnt)
#define USE_ASYNC_LDS 1
#endif
#endif

#define AS_GLOBAL __attribute__((address_space(1)))
#define AS_LDS    __attribute__((address_space(3)))

#if USE_ASYNC_LDS
// Builtin param types (from the toolchain's diagnostic): GNU-vector int4
// pointers, non-const, AS1 source / AS3 destination.
typedef int v4i_gnu __attribute__((vector_size(16)));
typedef AS_GLOBAL v4i_gnu* async_gptr;
typedef AS_LDS    v4i_gnu* async_lptr;

__device__ __forceinline__ void async_copy_b128(const _Float16* g, _Float16* l) {
  __builtin_amdgcn_global_load_async_to_lds_b128(
      (async_gptr)(g), (async_lptr)(l), 0, 0);
}
#endif

// ---------------------------------------------------------------------------
// Pass 1: dequantize weights. w_f16[n,k] = (f16)(w_int8[n,k] * scale[n]).
// ---------------------------------------------------------------------------
__global__ __launch_bounds__(256) void dequant_weights_kernel(
    const int* __restrict__ w, const float* __restrict__ scale,
    _Float16* __restrict__ out) {
  size_t v = (size_t)blockIdx.x * 256u + threadIdx.x;   // vec4 index
  unsigned n = (unsigned)(v >> 10);                     // / (K_TOT/4)
  v4i wi = ((const v4i*)w)[v];
  float s = scale[n];
  v4h r;
  r[0] = (_Float16)((float)wi[0] * s);
  r[1] = (_Float16)((float)wi[1] * s);
  r[2] = (_Float16)((float)wi[2] * s);
  r[3] = (_Float16)((float)wi[3] * s);
  ((v4h*)out)[v] = r;
}

// ---------------------------------------------------------------------------
// Pass 2: x fp32 -> f16.
// ---------------------------------------------------------------------------
__global__ __launch_bounds__(256) void convert_x_kernel(
    const float* __restrict__ x, _Float16* __restrict__ out) {
  size_t v = (size_t)blockIdx.x * 256u + threadIdx.x;   // vec4 index
  v4f xi = ((const v4f*)x)[v];
  v4h r;
  r[0] = (_Float16)xi[0];
  r[1] = (_Float16)xi[1];
  r[2] = (_Float16)xi[2];
  r[3] = (_Float16)xi[3];
  ((v4h*)out)[v] = r;
}

// ---------------------------------------------------------------------------
// Pass 3: WMMA GEMM. out[m,n] = sum_k A16[m,k]*W16[n,k] + bias[n].
// Block: 256 threads (8 wave32s), tile 128(M) x 128(N), K-step 32.
// Wave grid 2(M) x 4(N): each wave owns 64x32 = 4x2 WMMA 16x16 tiles.
// ---------------------------------------------------------------------------
__global__ __launch_bounds__(256) void gemm_wmma_kernel(
    const _Float16* __restrict__ A,   // [M_TOT, K_TOT] f16
    const _Float16* __restrict__ Bw,  // [N_TOT, K_TOT] f16 (dequantized W)
    const float* __restrict__ bias,   // [N_TOT]
    float* __restrict__ out) {        // [M_TOT, N_TOT]
  __shared__ alignas(16) _Float16 As[2][BLK_M * LDSROW];
  __shared__ alignas(16) _Float16 Bs[2][BLK_M * LDSROW];

  const int tid   = threadIdx.x;
  const int lane  = tid & 31;
  const int wave  = tid >> 5;
  const int wm    = wave >> 2;    // 0..1  (M direction)
  const int wn    = wave & 3;     // 0..3  (N direction)
  const int lrow  = lane & 15;
  const int lhalf = lane >> 4;

  const int bm = blockIdx.y * BLK_M;
  const int bn = blockIdx.x * BLK_N;

  // Global->LDS staging: each thread moves 2x16B for A and 2x16B for B per K-step.
  const int srow   = tid >> 2;          // 0..63 (rows 0..63; +64 for second pass)
  const int schunk = (tid & 3) * 8;     // f16 offset within a 32-f16 row
  const _Float16* aG = A  + (size_t)(bm + srow) * K_TOT + schunk;
  const _Float16* bG = Bw + (size_t)(bn + srow) * K_TOT + schunk;
  const size_t rowStep = (size_t)64 * K_TOT;
  const int l0 = srow * LDSROW + schunk;

#if USE_ASYNC_LDS
  // ---- prologue: async-stage K-block 0 into buffer 0 (no VGPR round-trip)
  async_copy_b128(aG,           &As[0][l0]);
  async_copy_b128(aG + rowStep, &As[0][l0 + 64 * LDSROW]);
  async_copy_b128(bG,           &Bs[0][l0]);
  async_copy_b128(bG + rowStep, &Bs[0][l0 + 64 * LDSROW]);
  __builtin_amdgcn_s_wait_asynccnt(0);
  __syncthreads();
#else
  // ---- prologue: register-staged K-block 0 into buffer 0
  {
    v8h ra0 = *(const v8h*)(aG);
    v8h ra1 = *(const v8h*)(aG + rowStep);
    v8h rb0 = *(const v8h*)(bG);
    v8h rb1 = *(const v8h*)(bG + rowStep);
    *(v8h*)&As[0][l0]               = ra0;
    *(v8h*)&As[0][l0 + 64 * LDSROW] = ra1;
    *(v8h*)&Bs[0][l0]               = rb0;
    *(v8h*)&Bs[0][l0 + 64 * LDSROW] = rb1;
  }
  __syncthreads();
#endif

  v8f acc[4][2] = {};

  const int NK = K_TOT / BLK_K;   // 128
  for (int ks = 0; ks < NK; ++ks) {
    const int cur = ks & 1;

#if USE_ASYNC_LDS
    // Kick off async copy of the next K-block into the other buffer; it
    // streams global->LDS in the background of this iteration's WMMA work.
    if (ks + 1 < NK) {
      const int nb = (ks + 1) & 1;
      const int kb = (ks + 1) * BLK_K;
      async_copy_b128(aG + kb,           &As[nb][l0]);
      async_copy_b128(aG + kb + rowStep, &As[nb][l0 + 64 * LDSROW]);
      async_copy_b128(bG + kb,           &Bs[nb][l0]);
      async_copy_b128(bG + kb + rowStep, &Bs[nb][l0 + 64 * LDSROW]);
    }
#else
    v8h ra0, ra1, rb0, rb1;
    if (ks + 1 < NK) {
      const int kb = (ks + 1) * BLK_K;
      ra0 = *(const v8h*)(aG + kb);
      ra1 = *(const v8h*)(aG + kb + rowStep);
      rb0 = *(const v8h*)(bG + kb);
      rb1 = *(const v8h*)(bG + kb + rowStep);
    }
#endif

    // A fragments: lane<16 holds K {0..7, 16..23}; lane>=16 holds {8..15, 24..31}
    v16h af[4];
#pragma unroll
    for (int i = 0; i < 4; ++i) {
      const int base = (wm * 64 + i * 16 + lrow) * LDSROW + lhalf * 8;
      v8h lo = *(const v8h*)&As[cur][base];
      v8h hi = *(const v8h*)&As[cur][base + 16];
      af[i] = __builtin_shufflevector(lo, hi, 0, 1, 2, 3, 4, 5, 6, 7,
                                      8, 9, 10, 11, 12, 13, 14, 15);
    }
    // B fragments: lane<16 holds K {0..7, 16..23}; lane>=16 holds {8..15, 24..31}
    v16h bf[2];
#pragma unroll
    for (int j = 0; j < 2; ++j) {
      const int base = (wn * 32 + j * 16 + lrow) * LDSROW + lhalf * 8;
      v8h lo = *(const v8h*)&Bs[cur][base];
      v8h hi = *(const v8h*)&Bs[cur][base + 16];
      bf[j] = __builtin_shufflevector(lo, hi, 0, 1, 2, 3, 4, 5, 6, 7,
                                      8, 9, 10, 11, 12, 13, 14, 15);
    }

#pragma unroll
    for (int i = 0; i < 4; ++i)
#pragma unroll
      for (int j = 0; j < 2; ++j)
        acc[i][j] = __builtin_amdgcn_wmma_f32_16x16x32_f16(
            false, af[i], false, bf[j], (short)0, acc[i][j], false, false);

#if USE_ASYNC_LDS
    if (ks + 1 < NK) __builtin_amdgcn_s_wait_asynccnt(0);
#else
    if (ks + 1 < NK) {
      const int nb = (ks + 1) & 1;
      *(v8h*)&As[nb][l0]               = ra0;
      *(v8h*)&As[nb][l0 + 64 * LDSROW] = ra1;
      *(v8h*)&Bs[nb][l0]               = rb0;
      *(v8h*)&Bs[nb][l0 + 64 * LDSROW] = rb1;
    }
#endif
    __syncthreads();
  }

  // ---- epilogue: C layout is n = lane&15, m = r + 8*(lane>>4) per 16x16 tile
#pragma unroll
  for (int j = 0; j < 2; ++j) {
    const int n = bn + wn * 32 + j * 16 + lrow;
    const float bj = bias[n];
#pragma unroll
    for (int i = 0; i < 4; ++i) {
      const int mbase = bm + wm * 64 + i * 16 + lhalf * 8;
#pragma unroll
      for (int r = 0; r < 8; ++r) {
        out[(size_t)(mbase + r) * N_TOT + n] = acc[i][j][r] + bj;
      }
    }
  }
}

// ---------------------------------------------------------------------------
extern "C" void kernel_launch(void* const* d_in, const int* in_sizes, int n_in,
                              void* d_out, int out_size, void* d_ws, size_t ws_size,
                              hipStream_t stream) {
  const float* x     = (const float*)d_in[0];  // [4,2048,4096] fp32
  const int*   w     = (const int*)d_in[1];    // [11008,4096] int8-in-int32
  const float* scale = (const float*)d_in[2];  // [11008,1]
  const float* bias  = (const float*)d_in[3];  // [11008]
  float* out = (float*)d_out;                  // [4,2048,11008] fp32

  // Workspace layout: f16 weights (86MB) then f16 activations (64MB).
  _Float16* w16 = (_Float16*)d_ws;
  _Float16* x16 = (_Float16*)((char*)d_ws + (size_t)N_TOT * K_TOT * sizeof(_Float16));

  // Pass 1: dequant weights. N*K/4 vec4 elements -> 44,032 blocks.
  dequant_weights_kernel<<<(N_TOT * (size_t)K_TOT) / 4 / 256, 256, 0, stream>>>(
      w, scale, w16);

  // Pass 2: x fp32->f16. M*K/4 -> 32,768 blocks.
  convert_x_kernel<<<(M_TOT * (size_t)K_TOT) / 4 / 256, 256, 0, stream>>>(x, x16);

  // Pass 3: WMMA GEMM. Grid (N/128, M/128) = (86, 64).
  dim3 grid(N_TOT / BLK_N, M_TOT / BLK_M);
  gemm_wmma_kernel<<<grid, 256, 0, stream>>>(x16, w16, bias, out);
}
